// GatedAttentionBlock_33165737459846
// MI455X (gfx1250) — matrix operand order repaired and checked by reading, not compiled, not run on hardware
//
#include <hip/hip_runtime.h>
#include <hip/hip_bf16.h>

// ---------------------------------------------------------------------------
// Gated attention block on gfx1250 (MI455X): all heavy math through
// v_wmma_f32_16x16x32_bf16 (bf16 in, f32 accumulate).
// Sizes fixed by the reference: B=2, S=2048, D=1024, H=16, HD=64, BLOCK=128.
// ---------------------------------------------------------------------------

typedef unsigned short u16;
typedef __attribute__((ext_vector_type(16))) __bf16 v16bf;
typedef __attribute__((ext_vector_type(8)))  float  v8f;

#define DEV static __device__ __forceinline__

typedef struct U4_ { unsigned x, y, z, w; } __attribute__((aligned(16), may_alias)) U4;

union FragBF { U4 q[2]; unsigned u[8]; v16bf v; };
union FragF  { v8f v; float f[8]; };

DEV u16 f2bf(float f) {
  union { float f; unsigned u; } c; c.f = f;
  unsigned r = c.u + 0x7fffu + ((c.u >> 16) & 1u);   // round-to-nearest-even
  return (u16)(r >> 16);
}
DEV unsigned pack2(float a, float b) { return (unsigned)f2bf(a) | ((unsigned)f2bf(b) << 16); }

// A-matrix fragment (16x32 bf16): lane = M row; lanes 0-15 hold K k0..k0+15,
// lanes 16-31 hold K k0+16..k0+31. Source row-major [rows, ldr].
DEV v16bf load_fragA(const u16* p, int ldr, int row0, int k0) {
  int lane = threadIdx.x & 31;
  const u16* base = p + (size_t)(row0 + (lane & 15)) * ldr + k0 + ((lane >> 4) << 4);
  FragBF f;
  f.q[0] = *(const U4*)(base);
  f.q[1] = *(const U4*)(base + 8);
  return f.v;
}
// B-matrix fragment (32x16 bf16): lane = N col; lanes 0-15 hold K k0..k0+15,
// lanes 16-31 hold K k0+16..k0+31. Source = transposed matrix, row-major [N, ldr].
DEV v16bf load_fragB(const u16* p, int ldr, int col0, int k0) {
  int lane = threadIdx.x & 31;
  const u16* base = p + (size_t)(col0 + (lane & 15)) * ldr + k0 + ((lane >> 4) << 4);
  FragBF f;
  f.q[0] = *(const U4*)(base);
  f.q[1] = *(const U4*)(base + 8);
  return f.v;
}
DEV v8f wmma_bf(v16bf a, v16bf b, v8f c) {
  return __builtin_amdgcn_wmma_f32_16x16x32_bf16(false, a, false, b, (short)0, c, false, false);
}
DEV v8f vzero() { v8f z = {0.f,0.f,0.f,0.f,0.f,0.f,0.f,0.f}; return z; }

// Per-wave 16xM0 x 64-col GEMM core: C += A[m0:,:K] * BT[n0:,:K]^T
DEV void gemm4(const u16* A, const u16* BT, int K, int m0, int n0, v8f acc[4]) {
  for (int k0 = 0; k0 < K; k0 += 32) {
    v16bf a = load_fragA(A, K, m0, k0);
#pragma unroll
    for (int t = 0; t < 4; ++t)
      acc[t] = wmma_bf(a, load_fragB(BT, K, n0 + 16 * t, k0), acc[t]);
  }
}

// --------------------------- small prep kernels ----------------------------

__global__ void k_householder(const float* __restrict__ vs, float* __restrict__ Qout) {
  __shared__ float Qm[64 * 64];
  __shared__ float vsh[64];
  __shared__ float wsh[64];
  __shared__ float vns;
  int t = threadIdx.x;
  for (int i = t; i < 4096; i += 256) Qm[i] = ((i >> 6) == (i & 63)) ? 1.f : 0.f;
  __syncthreads();
  for (int r = 0; r < 32; ++r) {
    if (t < 64) vsh[t] = vs[r * 64 + t];
    __syncthreads();
    if (t == 0) { float s = 0.f; for (int i = 0; i < 64; ++i) s += vsh[i] * vsh[i]; vns = 2.0f / (s + 1e-8f); }
    if (t < 64) { float s = 0.f; for (int i = 0; i < 64; ++i) s += vsh[i] * Qm[i * 64 + t]; wsh[t] = s; }
    __syncthreads();
    float c = vns;
    for (int i = t; i < 4096; i += 256) Qm[i] -= c * vsh[i >> 6] * wsh[i & 63];
    __syncthreads();
  }
  for (int i = t; i < 4096; i += 256) Qout[i] = Qm[i];
}

__global__ void k_rope(float* __restrict__ cosb, float* __restrict__ sinb) {
  int i = blockIdx.x * 256 + threadIdx.x;
  if (i >= 2048 * 64) return;
  int s = i >> 6, d = i & 63, j = d & 31;
  float inv = powf(10000.0f, -(float)(2 * j) / 64.0f);
  float a = (float)s * inv;
  cosb[i] = cosf(a);
  sinb[i] = sinf(a);
}

// wT[n][k] = bf16(w[k][n]); w is [K,N] row-major
__global__ void k_castT(const float* __restrict__ w, u16* __restrict__ wT, int K, int N) {
  size_t i = (size_t)blockIdx.x * 256 + threadIdx.x;
  if (i >= (size_t)K * N) return;
  int n = (int)(i / K), k = (int)(i % K);
  wT[i] = f2bf(w[(size_t)k * N + n]);
}

__global__ void k_rmsnorm(const float* __restrict__ x, u16* __restrict__ out) {
  int row = blockIdx.x;
  const float* xr = x + (size_t)row * 1024;
  float s = 0.f;
  for (int i = threadIdx.x; i < 1024; i += 256) { float v = xr[i]; s += v * v; }
  __shared__ float red[256];
  red[threadIdx.x] = s; __syncthreads();
  for (int o = 128; o > 0; o >>= 1) { if (threadIdx.x < o) red[threadIdx.x] += red[threadIdx.x + o]; __syncthreads(); }
  float rinv = rsqrtf(red[0] * (1.0f / 1024.0f) + 1e-6f);
  for (int i = threadIdx.x; i < 1024; i += 256) out[(size_t)row * 1024 + i] = f2bf(xr[i] * rinv);
}

// qkv -> per-head: t1 = t @ Q^T ; rope ; t2 = r @ Q ; store q/k bf16 [B,H,S,64],
// plus v transposed bf16 [B,H,64,S] for attention B/A fragments.
__global__ void k_rotrope(const float* __restrict__ qkv, const float* __restrict__ Qmat,
                          const float* __restrict__ cosb, const float* __restrict__ sinb,
                          u16* __restrict__ qb, u16* __restrict__ kb, u16* __restrict__ vtb) {
  __shared__ float Qs[4096];
  __shared__ float cs[64], sn[64];
  __shared__ float buf[4][64];
  __shared__ float buf2[4][64];
  int t = threadIdx.x;
  int bs = blockIdx.x;                 // b*2048 + s
  int b = bs >> 11, s = bs & 2047;
  for (int i = t; i < 4096; i += 256) Qs[i] = Qmat[i];
  if (t < 64) { cs[t] = cosb[s * 64 + t]; sn[t] = sinb[s * 64 + t]; }
  __syncthreads();
  int d = t & 63, vsel = t >> 6;       // 4 vectors in flight
  const float* base = qkv + (size_t)bs * 3072;
  for (int it = 0; it < 8; ++it) {
    int vi = it * 4 + vsel;            // 0..31 : (head, q/k)
    int head = vi >> 1, isk = vi & 1;
    buf[vsel][d] = base[isk * 1024 + head * 64 + d];
    __syncthreads();
    float t1 = 0.f;                    // (t @ Q^T)[d] = sum_i t[i]*Q[d][i]
    for (int i = 0; i < 64; ++i) t1 += buf[vsel][i] * Qs[d * 64 + i];
    buf2[vsel][d] = t1;
    __syncthreads();
    float rh = (d < 32) ? -buf2[vsel][d + 32] : buf2[vsel][d - 32];
    float r = t1 * cs[d] + rh * sn[d];
    buf[vsel][d] = r;
    __syncthreads();
    float t2 = 0.f;                    // (r @ Q)[d] = sum_i r[i]*Q[i][d]
    for (int i = 0; i < 64; ++i) t2 += buf[vsel][i] * Qs[i * 64 + d];
    u16 o = f2bf(t2);
    size_t dst = ((size_t)(b * 16 + head) * 2048 + s) * 64 + d;
    if (isk) kb[dst] = o; else qb[dst] = o;
    __syncthreads();
  }
  for (int i = t; i < 1024; i += 256) {
    int head = i >> 6, dd = i & 63;
    vtb[((size_t)(b * 16 + head) * 64 + dd) * 2048 + s] = f2bf(base[2048 + i]);
  }
}

// ----------------------------- flash attention -----------------------------
// One wave = 16 query rows of one (b,h). Computes S^T = K*Q^T (D-tile lane = q,
// matching softmax stats), then O^T = V^T*P^T; P^T B-frags built in-register
// with one xor-16 half-exchange. Block-causal @128 => loop bound only, no mask.
__global__ void __launch_bounds__(256) k_attn(const u16* __restrict__ qb, const u16* __restrict__ kb,
                                              const u16* __restrict__ vtb, u16* __restrict__ attn) {
  int w = blockIdx.x * 8 + (threadIdx.x >> 5);
  int qt = w & 127;
  int hh = (w >> 7) & 15;
  int b  = w >> 11;
  const u16* qp = qb  + (size_t)(b * 16 + hh) * 2048 * 64;
  const u16* kp = kb  + (size_t)(b * 16 + hh) * 2048 * 64;
  const u16* vp = vtb + (size_t)(b * 16 + hh) * 64 * 2048;
  int q0 = qt * 16;
  int kmax = ((q0 >> 7) + 1) << 7;                   // block-causal visibility
  v16bf Qb0 = load_fragB(qp, 64, q0, 0);             // B = Q^T (d 0..31)
  v16bf Qb1 = load_fragB(qp, 64, q0, 32);            // B = Q^T (d 32..63)
  v8f o[4];
#pragma unroll
  for (int t = 0; t < 4; ++t) o[t] = vzero();
  float m = -1e30f, l = 0.f;
  int lane = threadIdx.x & 31;
  bool hi = lane >= 16;
  for (int kk = 0; kk < kmax; kk += 32) {
    v8f s0 = vzero(), s1 = vzero();
    s0 = wmma_bf(load_fragA(kp, 64, kk,      0),  Qb0, s0);
    s0 = wmma_bf(load_fragA(kp, 64, kk,      32), Qb1, s0);
    s1 = wmma_bf(load_fragA(kp, 64, kk + 16, 0),  Qb0, s1);
    s1 = wmma_bf(load_fragA(kp, 64, kk + 16, 32), Qb1, s1);
    FragF f0, f1; f0.v = s0; f1.v = s1;
    float mx = -1e30f;
#pragma unroll
    for (int j = 0; j < 8; ++j) {
      f0.f[j] *= 0.125f; f1.f[j] *= 0.125f;          // 1/sqrt(64)
      mx = fmaxf(mx, fmaxf(f0.f[j], f1.f[j]));
    }
    mx = fmaxf(mx, __shfl_xor(mx, 16));
    float mn = fmaxf(m, mx);
    float alpha = __expf(m - mn);
    float ps = 0.f;
#pragma unroll
    for (int j = 0; j < 8; ++j) {
      f0.f[j] = __expf(f0.f[j] - mn);
      f1.f[j] = __expf(f1.f[j] - mn);
      ps += f0.f[j] + f1.f[j];
    }
    ps += __shfl_xor(ps, 16);
    l = l * alpha + ps;
    m = mn;
    // Build P^T B-fragment: lanes<16 need kk+0..15 (own even f + partner even f),
    // lanes>=16 need kk+16..31 (partner odd f + own odd f).
    float oe[8], oo[8];
#pragma unroll
    for (int j = 0; j < 8; ++j) { oe[j] = __shfl_xor(f0.f[j], 16); oo[j] = __shfl_xor(f1.f[j], 16); }
    FragBF pf;
#pragma unroll
    for (int j = 0; j < 4; ++j) {
      float a0 = hi ? oo[2*j]     : f0.f[2*j];
      float a1 = hi ? oo[2*j + 1] : f0.f[2*j + 1];
      float b0 = hi ? f1.f[2*j]     : oe[2*j];
      float b1 = hi ? f1.f[2*j + 1] : oe[2*j + 1];
      pf.u[j]     = pack2(a0, a1);
      pf.u[4 + j] = pack2(b0, b1);
    }
#pragma unroll
    for (int t = 0; t < 4; ++t) {
      o[t] = o[t] * alpha;
      o[t] = wmma_bf(load_fragA(vp, 2048, 16 * t, kk), pf.v, o[t]);  // A = V^T rows d
    }
  }
  float rl = 1.f / l;
  int r = lane & 15, hf = lane >> 4;
  size_t rowbase = (size_t)(b * 2048 + q0 + r) * 1024;
#pragma unroll
  for (int t = 0; t < 4; ++t) {
    FragF of; of.v = o[t];
    U4 st;
    st.x = pack2(of.f[0] * rl, of.f[1] * rl);
    st.y = pack2(of.f[2] * rl, of.f[3] * rl);
    st.z = pack2(of.f[4] * rl, of.f[5] * rl);
    st.w = pack2(of.f[6] * rl, of.f[7] * rl);
    *(U4*)(attn + rowbase + hh * 64 + 16 * t + 8 * hf) = st;
  }
}

// ------------------------------- GEMM kernels ------------------------------

__global__ void __launch_bounds__(256) k_gemm_qkv(const u16* __restrict__ A, const u16* __restrict__ BT,
                                                  float* __restrict__ C) {
  int w = blockIdx.x * 8 + (threadIdx.x >> 5);
  int nt = w % 48, mt = w / 48;
  int m0 = mt * 16, n0 = nt * 64;
  v8f acc[4]; for (int t = 0; t < 4; ++t) acc[t] = vzero();
  gemm4(A, BT, 1024, m0, n0, acc);
  int lane = threadIdx.x & 31, r = lane & 15, hf = lane >> 4;
#pragma unroll
  for (int t = 0; t < 4; ++t) {
    FragF f; f.v = acc[t];
#pragma unroll
    for (int j = 0; j < 8; ++j)
      C[(size_t)(m0 + 8 * hf + j) * 3072 + n0 + 16 * t + r] = f.f[j];
  }
}

__global__ void __launch_bounds__(256) k_gemm_outproj(const u16* __restrict__ A, const u16* __restrict__ BT,
                                                      float* __restrict__ C, u16* __restrict__ Cbf) {
  int w = blockIdx.x * 8 + (threadIdx.x >> 5);
  int nt = w & 15, mt = w >> 4;
  int m0 = mt * 16, n0 = nt * 64;
  v8f acc[4]; for (int t = 0; t < 4; ++t) acc[t] = vzero();
  gemm4(A, BT, 1024, m0, n0, acc);
  int lane = threadIdx.x & 31, r = lane & 15, hf = lane >> 4;
#pragma unroll
  for (int t = 0; t < 4; ++t) {
    FragF f; f.v = acc[t];
#pragma unroll
    for (int j = 0; j < 8; ++j) {
      size_t idx = (size_t)(m0 + 8 * hf + j) * 1024 + n0 + 16 * t + r;
      C[idx] = f.f[j];
      Cbf[idx] = f2bf(f.f[j]);
    }
  }
}

__global__ void __launch_bounds__(256) k_gemm_gate(const u16* __restrict__ A, const u16* __restrict__ BT,
                                                   const float* __restrict__ bgate,
                                                   const float* __restrict__ aout,
                                                   const float* __restrict__ xin,
                                                   float* __restrict__ xmid) {
  int w = blockIdx.x * 8 + (threadIdx.x >> 5);
  int nt = w & 15, mt = w >> 4;
  int m0 = mt * 16, n0 = nt * 64;
  v8f acc[4]; for (int t = 0; t < 4; ++t) acc[t] = vzero();
  gemm4(A, BT, 1024, m0, n0, acc);
  int lane = threadIdx.x & 31, r = lane & 15, hf = lane >> 4;
#pragma unroll
  for (int t = 0; t < 4; ++t) {
    FragF f; f.v = acc[t];
    int col = n0 + 16 * t + r;
    float bb = bgate[col];
#pragma unroll
    for (int j = 0; j < 8; ++j) {
      size_t idx = (size_t)(m0 + 8 * hf + j) * 1024 + col;
      float sg = 1.f / (1.f + __expf(-(f.f[j] + bb)));
      xmid[idx] = xin[idx] + aout[idx] * sg;
    }
  }
}

// MLP gate: same wave computes matching g (cols n0) and val (cols n0+4096)
// tiles so act = val*gelu(g) is fused without materializing gv.
__global__ void __launch_bounds__(256) k_gemm_mlp(const u16* __restrict__ A, const u16* __restrict__ BT,
                                                  const float* __restrict__ bias, u16* __restrict__ act) {
  int w = blockIdx.x * 8 + (threadIdx.x >> 5);
  int nt = w & 63, mt = w >> 6;
  int m0 = mt * 16, n0 = nt * 64;
  v8f accg[4], accv[4];
  for (int t = 0; t < 4; ++t) { accg[t] = vzero(); accv[t] = vzero(); }
  for (int k0 = 0; k0 < 1024; k0 += 32) {
    v16bf a = load_fragA(A, 1024, m0, k0);
#pragma unroll
    for (int t = 0; t < 4; ++t) {
      accg[t] = wmma_bf(a, load_fragB(BT, 1024, n0 + 16 * t, k0), accg[t]);
      accv[t] = wmma_bf(a, load_fragB(BT, 1024, 4096 + n0 + 16 * t, k0), accv[t]);
    }
  }
  int lane = threadIdx.x & 31, r = lane & 15, hf = lane >> 4;
#pragma unroll
  for (int t = 0; t < 4; ++t) {
    FragF fg, fv; fg.v = accg[t]; fv.v = accv[t];
    int col = n0 + 16 * t + r;
    float bg = bias[col], bv = bias[4096 + col];
#pragma unroll
    for (int j = 0; j < 8; ++j) {
      float gg = fg.f[j] + bg;
      float vv = fv.f[j] + bv;
      float ge = 0.5f * gg * (1.f + erff(gg * 0.70710678118f));   // exact GELU
      act[(size_t)(m0 + 8 * hf + j) * 4096 + col] = f2bf(vv * ge);
    }
  }
}

__global__ void __launch_bounds__(256) k_gemm_mlpout(const u16* __restrict__ A, const u16* __restrict__ BT,
                                                     const float* __restrict__ bout,
                                                     const float* __restrict__ xmid,
                                                     float* __restrict__ out) {
  int w = blockIdx.x * 8 + (threadIdx.x >> 5);
  int nt = w & 15, mt = w >> 4;
  int m0 = mt * 16, n0 = nt * 64;
  v8f acc[4]; for (int t = 0; t < 4; ++t) acc[t] = vzero();
  gemm4(A, BT, 4096, m0, n0, acc);
  int lane = threadIdx.x & 31, r = lane & 15, hf = lane >> 4;
#pragma unroll
  for (int t = 0; t < 4; ++t) {
    FragF f; f.v = acc[t];
    int col = n0 + 16 * t + r;
    float bb = bout[col];
#pragma unroll
    for (int j = 0; j < 8; ++j) {
      size_t idx = (size_t)(m0 + 8 * hf + j) * 1024 + col;
      out[idx] = xmid[idx] + f.f[j] + bb;
    }
  }
}

// --------------------------------- launch ----------------------------------

extern "C" void kernel_launch(void* const* d_in, const int* in_sizes, int n_in,
                              void* d_out, int out_size, void* d_ws, size_t ws_size,
                              hipStream_t stream) {
  (void)in_sizes; (void)n_in; (void)out_size; (void)ws_size;
  const float* x          = (const float*)d_in[0];
  // d_in[1]=mask (block-causal structure known), d_in[2]=grid_size : unused
  const float* vs         = (const float*)d_in[3];
  const float* w_qkv      = (const float*)d_in[4];
  const float* w_out      = (const float*)d_in[5];
  const float* w_gate     = (const float*)d_in[6];
  const float* b_gate     = (const float*)d_in[7];
  const float* w_mlp_gate = (const float*)d_in[8];
  const float* b_mlp_gate = (const float*)d_in[9];
  const float* w_mlp_out  = (const float*)d_in[10];
  const float* b_mlp_out  = (const float*)d_in[11];
  float* out = (float*)d_out;

  char* p = (char*)d_ws;
  auto alloc = [&](size_t bytes) { char* r = p; p += (bytes + 255) & ~(size_t)255; return (void*)r; };

  float* Qm      = (float*)alloc(4096 * 4);
  float* cosb    = (float*)alloc((size_t)2048 * 64 * 4);
  float* sinb    = (float*)alloc((size_t)2048 * 64 * 4);
  u16*   wT_qkv  = (u16*)alloc((size_t)3072 * 1024 * 2);
  u16*   wT_out  = (u16*)alloc((size_t)1024 * 1024 * 2);
  u16*   wT_gate = (u16*)alloc((size_t)1024 * 1024 * 2);
  u16*   wT_mlpg = (u16*)alloc((size_t)8192 * 1024 * 2);
  u16*   wT_mlpo = (u16*)alloc((size_t)1024 * 4096 * 2);
  u16*   hbuf    = (u16*)alloc((size_t)4096 * 1024 * 2);
  float* qkvbuf  = (float*)alloc((size_t)4096 * 3072 * 4);
  u16*   qbuf    = (u16*)alloc((size_t)32 * 2048 * 64 * 2);
  u16*   kbuf    = (u16*)alloc((size_t)32 * 2048 * 64 * 2);
  u16*   vtbuf   = (u16*)alloc((size_t)32 * 64 * 2048 * 2);
  u16*   attnb   = (u16*)alloc((size_t)4096 * 1024 * 2);
  float* aout    = (float*)alloc((size_t)4096 * 1024 * 4);
  u16*   aout_bf = (u16*)alloc((size_t)4096 * 1024 * 2);
  float* xmid    = (float*)alloc((size_t)4096 * 1024 * 4);
  u16*   h2buf   = (u16*)alloc((size_t)4096 * 1024 * 2);
  u16*   actb    = (u16*)alloc((size_t)4096 * 4096 * 2);

  k_householder<<<1, 256, 0, stream>>>(vs, Qm);
  k_rope<<<(2048 * 64 + 255) / 256, 256, 0, stream>>>(cosb, sinb);
  k_castT<<<(3072 * 1024) / 256, 256, 0, stream>>>(w_qkv, wT_qkv, 1024, 3072);
  k_castT<<<(1024 * 1024) / 256, 256, 0, stream>>>(w_out, wT_out, 1024, 1024);
  k_castT<<<(1024 * 1024) / 256, 256, 0, stream>>>(w_gate, wT_gate, 1024, 1024);
  k_castT<<<(8192 * 1024) / 256, 256, 0, stream>>>(w_mlp_gate, wT_mlpg, 1024, 8192);
  k_castT<<<(1024 * 4096) / 256, 256, 0, stream>>>(w_mlp_out, wT_mlpo, 4096, 1024);

  k_rmsnorm<<<4096, 256, 0, stream>>>(x, hbuf);
  k_gemm_qkv<<<(256 * 48) / 8, 256, 0, stream>>>(hbuf, wT_qkv, qkvbuf);        // 1536 blocks
  k_rotrope<<<4096, 256, 0, stream>>>(qkvbuf, Qm, cosb, sinb, qbuf, kbuf, vtbuf);
  k_attn<<<4096 / 8, 256, 0, stream>>>(qbuf, kbuf, vtbuf, attnb);              // 512 blocks
  k_gemm_outproj<<<(256 * 16) / 8, 256, 0, stream>>>(attnb, wT_out, aout, aout_bf);
  k_gemm_gate<<<(256 * 16) / 8, 256, 0, stream>>>(aout_bf, wT_gate, b_gate, aout, x, xmid);
  k_rmsnorm<<<4096, 256, 0, stream>>>(xmid, h2buf);
  k_gemm_mlp<<<(256 * 64) / 8, 256, 0, stream>>>(h2buf, wT_mlpg, b_mlp_gate, actb);  // 2048 blocks
  k_gemm_mlpout<<<(256 * 16) / 8, 256, 0, stream>>>(actb, wT_mlpo, b_mlp_out, xmid, out);
}
